// GAT_7851200217746
// MI455X (gfx1250) — matrix operand order repaired and checked by reading, not compiled, not run on hardware
//
#include <hip/hip_runtime.h>

// ---------------------------------------------------------------------------
// GAT hierarchy on gfx1250 (MI455X), wave32 + WMMA.
//
// Algebraic refactor: fold attention vectors into W (e = x·(W aᵀ)), softmax,
// xbar[h] = Σ_e alpha[e,h] x_e, out[h] = xbar[h] @ W[:,h*128:(h+1)*128].
// All GEMMs use v_wmma_f32_16x16x32_f16 with the split-f16 (hi+lo) trick for
// ~f32 accuracy: D = Ahi*Bhi + Ahi*Blo + Alo*Bhi (f32 accumulate).
//
// Fragment feeding is vectorized:
//  - B (weights) pre-swizzled into fragment order -> one v16h (2x b128) per
//    lane per tile, coalesced across the wave.
//  - A (xbar) written to LDS in fragment order -> one v16h (2x ds b128).
// ---------------------------------------------------------------------------

typedef __attribute__((ext_vector_type(16))) _Float16 v16h;
typedef __attribute__((ext_vector_type(8)))  float    v8f;

#define LEAKY_SLOPE 0.2f

__device__ __forceinline__ v8f wmma_f16(v16h a, v16h b, v8f c) {
    // (neg_a, A, neg_b, B, c_mod, C, reuse_a, reuse_b)
    return __builtin_amdgcn_wmma_f32_16x16x32_f16(false, a, false, b,
                                                  (short)0, c, false, false);
}

// 16-bit A 16x32 layout inversion: feature index f (K) -> (lane_half, elem)
//   lane = M + 16*((f>>3)&1),  elem = (f&7) + 8*(f>>4)
// 16-bit B 32x16 layout: lane: N=L&15, half=L>>4; elem e -> K = 8*half + (e&7) + 16*(e>>3).

// ---------------------------------------------------------------------------
// Prep kernels
// ---------------------------------------------------------------------------

// Pack W [F, N] (f32) into WMMA B-fragment order, split into f16 hi + lo.
// Fragment (kc, nt) occupies 512 halves: [lane][elem] with
//   K = kc + 8*(lane>>4) + (elem&7) + 16*(elem>>3),  col = nt*16 + (lane&15).
__global__ void k_packB(const float* __restrict__ W, _Float16* __restrict__ hip,
                        _Float16* __restrict__ lop, int Ncols, int total) {
    int idx = blockIdx.x * blockDim.x + threadIdx.x;
    if (idx < total) {
        int frag = idx >> 9;
        int r    = idx & 511;
        int lane = r >> 4;
        int elem = r & 15;
        int ntiles = Ncols >> 4;
        int kc = (frag / ntiles) << 5;
        int nt = frag - (frag / ntiles) * ntiles;
        int K  = kc + ((lane >> 4) << 3) + (elem & 7) + ((elem >> 3) << 4);
        int col = (nt << 4) + (lane & 15);
        float w = W[(size_t)K * Ncols + col];
        _Float16 h = (_Float16)w;
        hip[idx] = h;
        lop[idx] = (_Float16)(w - (float)h);
    }
}

// out[f*4+h] = sum_d W[f*512 + h*128 + d] * a[h*128 + d]   (fold a into W)
__global__ void k_fold(const float* __restrict__ W, const float* __restrict__ a,
                       float* __restrict__ out, int F) {
    int i = blockIdx.x * blockDim.x + threadIdx.x;
    if (i < F * 4) {
        int f = i >> 2, h = i & 3;
        const float* wp = W + (size_t)f * 512 + h * 128;
        const float* ap = a + h * 128;
        float s = 0.f;
        for (int d = 0; d < 128; ++d) s += wp[d] * ap[d];
        out[i] = s;
    }
}

// ---------------------------------------------------------------------------
// Fused GAT layer.  Xs: [G,F] self rows, Xn: [G*E,F] neighbor rows,
// Bhi/Blo: W [F,512] pre-swizzled fragments, Wn/Ws: [F,4] folded attn,
// Out: [G,512].  Block = 128 threads (4 waves); 16 groups; wave w -> head w.
// ---------------------------------------------------------------------------
template <int E, int F>
__global__ __launch_bounds__(128) void gat_fused(
    const float* __restrict__ Xs, const float* __restrict__ Xn,
    const _Float16* __restrict__ Bhip, const _Float16* __restrict__ Blop,
    const float* __restrict__ Wn, const float* __restrict__ Ws,
    float* __restrict__ Out) {
    constexpr int GT = 16;
    const int tid   = threadIdx.x;
    const int gbase = blockIdx.x * GT;

    __shared__ float s_w[2 * F * 4];        // folded Wn then Ws
    __shared__ float s_e[GT][E][4];         // logits -> alpha
    __shared__ float s_es[GT][4];
    __shared__ __align__(32) _Float16 s_ahi[4][32][16];  // A frags [head][lane][elem]
    __shared__ __align__(32) _Float16 s_alo[4][32][16];

    for (int i = tid; i < F * 4; i += 128) {
        s_w[i]         = Wn[i];
        s_w[F * 4 + i] = Ws[i];
    }
    __syncthreads();

    // ---- Phase 1: e_n[g,e,h], e_s[g,h] via folded dot products ----
    for (int r = tid; r < GT * (E + 1); r += 128) {
        int g = r / (E + 1);
        int e = r - g * (E + 1);
        const float* xp = (e < E) ? (Xn + ((size_t)(gbase + g) * E + e) * F)
                                  : (Xs + (size_t)(gbase + g) * F);
        const float* wv = (e < E) ? s_w : (s_w + F * 4);
        float a0 = 0.f, a1 = 0.f, a2 = 0.f, a3 = 0.f;
        for (int f = 0; f < F; f += 4) {
            float4 x = *reinterpret_cast<const float4*>(xp + f);
            const float* wp = wv + f * 4;
            a0 += x.x * wp[0] + x.y * wp[4] + x.z * wp[8]  + x.w * wp[12];
            a1 += x.x * wp[1] + x.y * wp[5] + x.z * wp[9]  + x.w * wp[13];
            a2 += x.x * wp[2] + x.y * wp[6] + x.z * wp[10] + x.w * wp[14];
            a3 += x.x * wp[3] + x.y * wp[7] + x.z * wp[11] + x.w * wp[15];
        }
        if (e < E) {
            s_e[g][e][0] = a0; s_e[g][e][1] = a1; s_e[g][e][2] = a2; s_e[g][e][3] = a3;
        } else {
            s_es[g][0] = a0; s_es[g][1] = a1; s_es[g][2] = a2; s_es[g][3] = a3;
        }
    }
    __syncthreads();

    // ---- Phase 2: leaky-relu + softmax over E, per (group, head) ----
    if (tid < GT * 4) {
        int g = tid >> 2, h = tid & 3;
        float es = s_es[g][h];
        float l[E];
        float m = -1e30f;
        for (int e = 0; e < E; ++e) {
            float v = es + s_e[g][e][h];
            v = (v > 0.f) ? v : LEAKY_SLOPE * v;
            l[e] = v;
            m = fmaxf(m, v);
        }
        float s = 0.f;
        for (int e = 0; e < E; ++e) { l[e] = expf(l[e] - m); s += l[e]; }
        float inv = 1.f / s;
        for (int e = 0; e < E; ++e) s_e[g][e][h] = l[e] * inv;  // alpha
    }
    __syncthreads();

    // ---- Phase 3: stream K in chunks of 32; xbar -> WMMA projection ----
    const int lane = tid & 31;
    const int head = tid >> 5;      // wave id == head
    const int half = lane >> 4;
    const int Ml   = lane & 15;
    v8f c[8] = {};

    for (int kc = 0; kc < F; kc += 32) {
        // (a) xbar chunk: xbar[g,h,f] = sum_e alpha[g,e,h] * Xn[g,e,kc+f]
        //     written to LDS directly in A-fragment order.
        for (int idx = tid; idx < GT * 32; idx += 128) {
            int g = idx >> 5, f = idx & 31;
            const float* xp = Xn + ((size_t)(gbase + g) * E) * F + kc + f;
            float a0 = 0.f, a1 = 0.f, a2 = 0.f, a3 = 0.f;
            for (int e = 0; e < E; ++e) {
                float x = xp[(size_t)e * F];
                a0 += s_e[g][e][0] * x;
                a1 += s_e[g][e][1] * x;
                a2 += s_e[g][e][2] * x;
                a3 += s_e[g][e][3] * x;
            }
            int alane = g + ((f >> 3) & 1) * 16;
            int aelem = (f & 7) + ((f >> 4) << 3);
            float acc[4] = {a0, a1, a2, a3};
#pragma unroll
            for (int h = 0; h < 4; ++h) {
                _Float16 hi = (_Float16)acc[h];
                s_ahi[h][alane][aelem] = hi;
                s_alo[h][alane][aelem] = (_Float16)(acc[h] - (float)hi);
            }
        }
        __syncthreads();

        // (b) fragments: contiguous vector loads
        v16h Ahi = *reinterpret_cast<const v16h*>(&s_ahi[head][lane][0]);
        v16h Alo = *reinterpret_cast<const v16h*>(&s_alo[head][lane][0]);
        const int fragbase = ((kc >> 5) << 5) + (head << 3);  // (kc/32)*32 + head*8
#pragma unroll
        for (int nt = 0; nt < 8; ++nt) {
            size_t off = ((size_t)(fragbase + nt) << 9) + (lane << 4);
            v16h Bhi = *reinterpret_cast<const v16h*>(Bhip + off);
            v16h Blo = *reinterpret_cast<const v16h*>(Blop + off);
            c[nt] = wmma_f16(Alo, Bhi, c[nt]);
            c[nt] = wmma_f16(Ahi, Blo, c[nt]);
            c[nt] = wmma_f16(Ahi, Bhi, c[nt]);
        }
        __syncthreads();   // protect s_ahi/s_alo before next chunk overwrites
    }

    // ---- Store: C/D layout: VGPR r -> row r + 8*half, col = lane&15 ----
#pragma unroll
    for (int nt = 0; nt < 8; ++nt) {
#pragma unroll
        for (int r = 0; r < 8; ++r) {
            int row = gbase + r + half * 8;
            int col = head * 128 + nt * 16 + Ml;
            Out[(size_t)row * 512 + col] = c[nt][r];
        }
    }
}

// ---------------------------------------------------------------------------
// Final FC: Out[1024,256] = A[1024,512] @ Wfc[512,256]  (split-f16 WMMA)
// Block = 128 threads; 16 rows per block; wave w covers cols [w*64, w*64+64).
// ---------------------------------------------------------------------------
__global__ __launch_bounds__(128) void fc_gemm(
    const float* __restrict__ A, const _Float16* __restrict__ Bhip,
    const _Float16* __restrict__ Blop, float* __restrict__ Out) {
    const int tid  = threadIdx.x;
    const int lane = tid & 31;
    const int w    = tid >> 5;
    const int half = lane >> 4;
    const int Ml   = lane & 15;
    const int m0   = blockIdx.x * 16;
    v8f c[4] = {};

    for (int kc = 0; kc < 512; kc += 32) {
        // A fragment: per-lane K pattern is two contiguous runs of 8 floats.
        const float* arow = A + (size_t)(m0 + Ml) * 512 + kc + 8 * half;
        float4 x0 = *reinterpret_cast<const float4*>(arow);
        float4 x1 = *reinterpret_cast<const float4*>(arow + 4);
        float4 y0 = *reinterpret_cast<const float4*>(arow + 16);
        float4 y1 = *reinterpret_cast<const float4*>(arow + 20);
        float av[16] = {x0.x, x0.y, x0.z, x0.w, x1.x, x1.y, x1.z, x1.w,
                        y0.x, y0.y, y0.z, y0.w, y1.x, y1.y, y1.z, y1.w};
        v16h Ahi, Alo;
#pragma unroll
        for (int e = 0; e < 16; ++e) {
            _Float16 h = (_Float16)av[e];
            Ahi[e] = h;
            Alo[e] = (_Float16)(av[e] - (float)h);
        }
        const int fragbase = ((kc >> 5) << 4) + (w << 2);  // (kc/32)*16 + w*4
#pragma unroll
        for (int nt = 0; nt < 4; ++nt) {
            size_t off = ((size_t)(fragbase + nt) << 9) + (lane << 4);
            v16h Bhi = *reinterpret_cast<const v16h*>(Bhip + off);
            v16h Blo = *reinterpret_cast<const v16h*>(Blop + off);
            c[nt] = wmma_f16(Alo, Bhi, c[nt]);
            c[nt] = wmma_f16(Ahi, Blo, c[nt]);
            c[nt] = wmma_f16(Ahi, Bhi, c[nt]);
        }
    }
#pragma unroll
    for (int nt = 0; nt < 4; ++nt) {
#pragma unroll
        for (int r = 0; r < 8; ++r) {
            int row = m0 + r + half * 8;
            int col = w * 64 + nt * 16 + Ml;
            Out[(size_t)row * 256 + col] = c[nt][r];
        }
    }
}

// ---------------------------------------------------------------------------
// Host launcher
// ---------------------------------------------------------------------------
extern "C" void kernel_launch(void* const* d_in, const int* in_sizes, int n_in,
                              void* d_out, int out_size, void* d_ws, size_t ws_size,
                              hipStream_t stream) {
    (void)in_sizes; (void)n_in; (void)out_size; (void)ws_size;
    const float* h0  = (const float*)d_in[0];   // [1024,128]
    const float* h1  = (const float*)d_in[1];   // [1024,10,128]
    const float* h2  = (const float*)d_in[2];   // [1024,250,128]
    const float* W0  = (const float*)d_in[3];   // [128,512]
    const float* a0s = (const float*)d_in[4];   // [4,128]
    const float* a0n = (const float*)d_in[5];
    const float* W1  = (const float*)d_in[6];   // [512,512]
    const float* a1s = (const float*)d_in[7];
    const float* a1n = (const float*)d_in[8];
    const float* Wfc = (const float*)d_in[9];   // [512,256]

    char* ws = (char*)d_ws;
    size_t off = 0;
    auto carve = [&](size_t bytes) -> char* {
        char* p = ws + off;
        off += (bytes + 255) & ~(size_t)255;
        return p;
    };
    _Float16* W0hi  = (_Float16*)carve((size_t)128 * 512 * 2);
    _Float16* W0lo  = (_Float16*)carve((size_t)128 * 512 * 2);
    _Float16* W1hi  = (_Float16*)carve((size_t)512 * 512 * 2);
    _Float16* W1lo  = (_Float16*)carve((size_t)512 * 512 * 2);
    _Float16* Wfchi = (_Float16*)carve((size_t)512 * 256 * 2);
    _Float16* Wfclo = (_Float16*)carve((size_t)512 * 256 * 2);
    float*    Wn0   = (float*)carve((size_t)128 * 4 * 4);
    float*    Ws0   = (float*)carve((size_t)128 * 4 * 4);
    float*    Wn1   = (float*)carve((size_t)512 * 4 * 4);
    float*    Ws1   = (float*)carve((size_t)512 * 4 * 4);
    float*    g1    = (float*)carve((size_t)10240 * 512 * 4);  // hs[1] after layer0
    float*    g0    = (float*)carve((size_t)1024 * 512 * 4);   // hs[0] after layer0
    float*    g0b   = (float*)carve((size_t)1024 * 512 * 4);   // hs[0] after layer1

    // Weight prep: swizzle into WMMA B-fragment order, split f16 hi/lo.
    k_packB<<<(128 * 512 + 255) / 256, 256, 0, stream>>>(W0, W0hi, W0lo, 512, 128 * 512);
    k_packB<<<(512 * 512 + 255) / 256, 256, 0, stream>>>(W1, W1hi, W1lo, 512, 512 * 512);
    k_packB<<<(512 * 256 + 255) / 256, 256, 0, stream>>>(Wfc, Wfchi, Wfclo, 256, 512 * 256);
    k_fold<<<(128 * 4 + 255) / 256, 256, 0, stream>>>(W0, a0n, Wn0, 128);
    k_fold<<<(128 * 4 + 255) / 256, 256, 0, stream>>>(W0, a0s, Ws0, 128);
    k_fold<<<(512 * 4 + 255) / 256, 256, 0, stream>>>(W1, a1n, Wn1, 512);
    k_fold<<<(512 * 4 + 255) / 256, 256, 0, stream>>>(W1, a1s, Ws1, 512);

    // Layer 0, j=1: self=h1 rows [10240,128], neigh=h2 [10240*25,128] -> g1
    gat_fused<25, 128><<<10240 / 16, 128, 0, stream>>>(h1, h2, W0hi, W0lo, Wn0, Ws0, g1);
    // Layer 0, j=0: self=h0 [1024,128], neigh=h1 [1024*10,128] -> g0
    gat_fused<10, 128><<<1024 / 16, 128, 0, stream>>>(h0, h1, W0hi, W0lo, Wn0, Ws0, g0);
    // Layer 1, j=0: self=g0 [1024,512], neigh=g1 [1024*10,512] -> g0b
    gat_fused<10, 512><<<1024 / 16, 128, 0, stream>>>(g0, g1, W1hi, W1lo, Wn1, Ws1, g0b);
    // Final FC
    fc_gemm<<<1024 / 16, 128, 0, stream>>>(g0b, Wfchi, Wfclo, (float*)d_out);
}
